// ARNN_71107478552688
// MI455X (gfx1250) — compile-verified, matrix operand order checked by reading, not run on hardware
//
#include <hip/hip_runtime.h>
#include <hip/hip_bf16.h>
#include <math.h>

typedef __attribute__((ext_vector_type(16))) _Float16 v16h;
typedef __attribute__((ext_vector_type(8)))  _Float16 v8h;
typedef __attribute__((ext_vector_type(8)))  float    v8f;

#define S_LEN 128
#define BATCH 64
#define IDIM  256
#define HDIM  512
#define ODIM  256
#define NMAX  8
#define KPAD  808            // 0:flag, 1..256:x, 257..263:0, 264..775:h, 776..807:0
                             // stride 808 f16 = 1616B -> 404%64=20 banks: conflict-free
#define KCH   25             // K-chunks of 32 for the hidden GEMM (covers k=0..799)
#define OKCH  (HDIM / 32)    // 16 K-chunks for the output GEMM
#define CLDS  3              // K-chunks of B-weights cached in LDS (swizzled)
#define WSTR  40             // LDS weight column slot: 40 f16 = 80B (conflict-free)

// ---------------------------------------------------------------------------
// Prep: build f16 weights pre-swizzled into WMMA B-fragment order.
// Wpack[(c*512 + j)*32 + k]  = Wc[j][32c + k]   (hidden GEMM, Wc = [flag|W_ih|W_hh])
// Wopack[(c*256 + j)*32 + k] = W_out[j][32c + k] (output GEMM)
// ---------------------------------------------------------------------------
__global__ void prep_weights(const float* __restrict__ W_ih,
                             const float* __restrict__ W_hh,
                             const float* __restrict__ W_out,
                             _Float16* __restrict__ Wpack,
                             _Float16* __restrict__ Wopack) {
  int idx = blockIdx.x * blockDim.x + threadIdx.x;
  const int NW = KCH * HDIM * 32;
  if (idx < NW) {
    int k = idx & 31;
    int j = (idx >> 5) & (HDIM - 1);
    int c = idx >> 14;           // /(32*512)
    int kk = c * 32 + k;
    float v = 0.f;
    if (kk == 0)                 v = W_ih[j * 257];            // ponder flag weight
    else if (kk <= 256)          v = W_ih[j * 257 + kk];       // input weights
    else if (kk >= 264 && kk < 776) v = W_hh[j * 512 + (kk - 264)]; // recurrent
    Wpack[idx] = (_Float16)v;
  } else {
    int id2 = idx - NW;
    if (id2 < OKCH * ODIM * 32) {
      int k = id2 & 31;
      int j = (id2 >> 5) & (ODIM - 1);
      int c = id2 >> 13;         // /(32*256)
      Wopack[id2] = (_Float16)W_out[j * 512 + c * 32 + k];
    }
  }
}

// ---------------------------------------------------------------------------
// Persistent ACT-RNN kernel: 4 workgroups, each owns 16 batch rows and runs
// the full 128-timestep x <=8-ponder-step chain with no global sync.
// ---------------------------------------------------------------------------
__launch_bounds__(256, 1)
__global__ void act_rnn(const float* __restrict__ x,
                        const float* __restrict__ s,
                        const float* __restrict__ b_ih,
                        const float* __restrict__ b_hh,
                        const float* __restrict__ b_out,
                        const float* __restrict__ W_halt,
                        const float* __restrict__ b_halt,
                        const _Float16* __restrict__ Wpack,
                        const _Float16* __restrict__ Wopack,
                        float* __restrict__ out,
                        float* __restrict__ pond_out) {
  __shared__ __align__(16) _Float16 Aext[16 * KPAD];           // ~25.9 KB
  __shared__ __align__(16) _Float16 Wlds[CLDS * HDIM * WSTR];  // ~120 KB, swizzled
  __shared__ float w_lds[16], cum[16], halt_acc[16], pond[16];
  __shared__ int   halted[16], allhalt;

  const int g      = blockIdx.x;         // batch slice: rows [16g, 16g+16)
  const int tid    = threadIdx.x;
  const int wave   = tid >> 5;
  const int lane   = tid & 31;
  const int lane_n = lane & 15;
  const int lane_hi = lane >> 4;
  const int koff   = lane_hi * 8;        // A/B fragment K sub-offset per half-wave
  const int nbase  = wave * 64;          // hidden columns [nbase, nbase+64) per wave

  // Per-lane constants (D-tile layout: n = lane_n, m = r + 8*lane_hi)
  float bias_h[4], whalt_l[4];
#pragma unroll
  for (int t4 = 0; t4 < 4; ++t4) {
    int j = nbase + 16 * t4 + lane_n;
    bias_h[t4]  = b_ih[j] + b_hh[j];
    whalt_l[t4] = W_halt[j];
  }
  float bias_o[2];
#pragma unroll
  for (int ot = 0; ot < 2; ++ot) bias_o[ot] = b_out[wave * 32 + 16 * ot + lane_n];
  const float bh = b_halt[0];

  // Cache first CLDS weight chunks into LDS, 80B column slots (bank-conflict-free).
  for (int i = tid; i < CLDS * HDIM * 32; i += 256) {
    int k = i & 31, cj = i >> 5;        // cj = c*512 + j
    Wlds[cj * WSTR + k] = Wpack[(size_t)cj * 32 + k];
  }
  // Init A: zero (pads + flag), then h0 from s.
  for (int i = tid; i < 16 * KPAD; i += 256) Aext[i] = (_Float16)0.f;
  __syncthreads();
  for (int i = tid; i < 16 * HDIM; i += 256) {
    int m = i >> 9, k = i & (HDIM - 1);
    Aext[m * KPAD + 264 + k] = (_Float16)s[(g * 16 + m) * HDIM + k];
  }
  __syncthreads();

  const v8f vzero = {};

  for (int t = 0; t < S_LEN; ++t) {
    // Load x_t slice into A columns 1..256; prefetch next timestep's slice.
    for (int i = tid; i < 16 * IDIM; i += 256) {
      int m = i >> 8, c = i & (IDIM - 1);
      Aext[m * KPAD + 1 + c] = (_Float16)x[((size_t)t * BATCH + g * 16 + m) * IDIM + c];
    }
    if (t + 1 < S_LEN && tid < 128)
      __builtin_prefetch(&x[((size_t)(t + 1) * BATCH + g * 16) * IDIM + tid * 32], 0, 1);
    if (tid < 16) { cum[tid] = 0.f; halted[tid] = 0; pond[tid] = 0.f; }

    v8f accs[4];                               // halting-weighted state acc_s
#pragma unroll
    for (int t4 = 0; t4 < 4; ++t4) accs[t4] = vzero;
    __syncthreads();

    for (int n = 0; n < NMAX; ++n) {
      if (tid < 16) { Aext[tid * KPAD] = (_Float16)((n == 0) ? 1.f : 0.f); halt_acc[tid] = 0.f; }
      if (tid == 0) allhalt = 1;
      __syncthreads();

      // ---- hidden GEMM: [16 x 800] x [800 x 512], wave owns 4 N-tiles ----
      v8f acc[4];
#pragma unroll
      for (int t4 = 0; t4 < 4; ++t4) acc[t4] = vzero;

      // chunks 0..CLDS-1: B fragments from LDS (swizzled, conflict-free)
      for (int c = 0; c < CLDS; ++c) {
        const _Float16* ap = &Aext[lane_n * KPAD + c * 32 + koff];
        v8h alo = *(const v8h*)(ap);
        v8h ahi = *(const v8h*)(ap + 16);
        v16h afrag = __builtin_shufflevector(alo, ahi,
            0, 1, 2, 3, 4, 5, 6, 7, 8, 9, 10, 11, 12, 13, 14, 15);
#pragma unroll
        for (int t4 = 0; t4 < 4; ++t4) {
          const _Float16* bp =
              &Wlds[(c * HDIM + nbase + 16 * t4 + lane_n) * WSTR + koff];
          v8h blo = *(const v8h*)(bp);
          v8h bhi = *(const v8h*)(bp + 16);
          v16h bfrag = __builtin_shufflevector(blo, bhi,
              0, 1, 2, 3, 4, 5, 6, 7, 8, 9, 10, 11, 12, 13, 14, 15);
          acc[t4] = __builtin_amdgcn_wmma_f32_16x16x32_f16(
              false, afrag, false, bfrag, (short)0, acc[t4], false, false);
        }
      }
      // chunks CLDS..24: B fragments streamed from L2-resident Wpack
      for (int c = CLDS; c < KCH; ++c) {
        const _Float16* ap = &Aext[lane_n * KPAD + c * 32 + koff];
        v8h alo = *(const v8h*)(ap);
        v8h ahi = *(const v8h*)(ap + 16);
        v16h afrag = __builtin_shufflevector(alo, ahi,
            0, 1, 2, 3, 4, 5, 6, 7, 8, 9, 10, 11, 12, 13, 14, 15);
        if (c + 1 < KCH)
          __builtin_prefetch(&Wpack[((size_t)((c + 1) * HDIM + nbase + lane_n)) * 32], 0, 3);
#pragma unroll
        for (int t4 = 0; t4 < 4; ++t4) {
          const _Float16* bp =
              &Wpack[((size_t)(c * HDIM + nbase + 16 * t4 + lane_n)) * 32 + koff];
          v8h blo = *(const v8h*)(bp);
          v8h bhi = *(const v8h*)(bp + 16);
          v16h bfrag = __builtin_shufflevector(blo, bhi,
              0, 1, 2, 3, 4, 5, 6, 7, 8, 9, 10, 11, 12, 13, 14, 15);
          acc[t4] = __builtin_amdgcn_wmma_f32_16x16x32_f16(
              false, afrag, false, bfrag, (short)0, acc[t4], false, false);
        }
      }

      // ---- bias + tanh in registers; per-lane halting partial dot ----
      float part[8];
#pragma unroll
      for (int r = 0; r < 8; ++r) part[r] = 0.f;
#pragma unroll
      for (int t4 = 0; t4 < 4; ++t4)
#pragma unroll
        for (int r = 0; r < 8; ++r) {
          float v = tanhf(acc[t4][r] + bias_h[t4]);
          acc[t4][r] = v;
          part[r] += v * whalt_l[t4];
        }
      // xor-shuffle tree over the 16 lanes of each half-wave, 1 atomic per row
#pragma unroll
      for (int r = 0; r < 8; ++r) {
        float v = part[r];
        v += __shfl_xor(v, 1);
        v += __shfl_xor(v, 2);
        v += __shfl_xor(v, 4);
        v += __shfl_xor(v, 8);
        if (lane_n == 0) atomicAdd(&halt_acc[r + 8 * lane_hi], v);
      }
      __syncthreads();

      // ---- halting state machine (16 scalar threads) ----
      if (tid < 16) {
        int m = tid;
        float p  = halted[m] ? 0.f : 1.f / (1.f + __expf(-(halt_acc[m] + bh)));
        float c0 = cum[m];
        bool ih  = (!halted[m]) && (((c0 + p) >= 1.f - 0.01f) || (n == NMAX - 1));
        w_lds[m] = ih ? (1.f - c0) : p;
        cum[m]   = c0 + p;
        if (ih) { halted[m] = 1; pond[m] = (1.f - c0) + (float)(n + 1); }
      }
      __syncthreads();

      // ---- acc_s += w*h_new ; publish h_new (f16) as next A ; all-halted? ----
#pragma unroll
      for (int t4 = 0; t4 < 4; ++t4)
#pragma unroll
        for (int r = 0; r < 8; ++r) {
          float w = w_lds[r + 8 * lane_hi];
          accs[t4][r] += w * acc[t4][r];
          Aext[(r + 8 * lane_hi) * KPAD + 264 + nbase + 16 * t4 + lane_n] =
              (_Float16)acc[t4][r];
        }
      if (tid < 16 && !halted[tid]) allhalt = 0;
      __syncthreads();
      int ah = allhalt;
      __syncthreads();            // protect allhalt reads from next-iter rewrite
      if (ah) break;              // uniform: all threads saw the same value
    }

    // ---- h_{t+1} = acc_s : publish weighted state as the new A h-region ----
#pragma unroll
    for (int t4 = 0; t4 < 4; ++t4)
#pragma unroll
      for (int r = 0; r < 8; ++r)
        Aext[(r + 8 * lane_hi) * KPAD + 264 + nbase + 16 * t4 + lane_n] =
            (_Float16)accs[t4][r];
    __syncthreads();

    // ---- output projection once per timestep: acc_o = acc_s @ W_out^T + b ----
    v8f oacc[2];
    oacc[0] = vzero; oacc[1] = vzero;
    for (int c = 0; c < OKCH; ++c) {
      const _Float16* ap = &Aext[lane_n * KPAD + 264 + c * 32 + koff];
      v8h alo = *(const v8h*)(ap);
      v8h ahi = *(const v8h*)(ap + 16);
      v16h afrag = __builtin_shufflevector(alo, ahi,
          0, 1, 2, 3, 4, 5, 6, 7, 8, 9, 10, 11, 12, 13, 14, 15);
#pragma unroll
      for (int ot = 0; ot < 2; ++ot) {
        const _Float16* bp =
            &Wopack[((size_t)(c * ODIM + wave * 32 + 16 * ot + lane_n)) * 32 + koff];
        v8h blo = *(const v8h*)(bp);
        v8h bhi = *(const v8h*)(bp + 16);
        v16h bfrag = __builtin_shufflevector(blo, bhi,
            0, 1, 2, 3, 4, 5, 6, 7, 8, 9, 10, 11, 12, 13, 14, 15);
        oacc[ot] = __builtin_amdgcn_wmma_f32_16x16x32_f16(
            false, afrag, false, bfrag, (short)0, oacc[ot], false, false);
      }
    }
#pragma unroll
    for (int ot = 0; ot < 2; ++ot)
#pragma unroll
      for (int r = 0; r < 8; ++r) {
        int m = r + 8 * lane_hi;
        out[((size_t)t * BATCH + g * 16 + m) * ODIM + wave * 32 + 16 * ot + lane_n] =
            oacc[ot][r] + bias_o[ot];
      }
    if (tid < 16) pond_out[t * BATCH + g * 16 + tid] = pond[tid];
    __syncthreads();
  }
}

// ---------------------------------------------------------------------------
extern "C" void kernel_launch(void* const* d_in, const int* in_sizes, int n_in,
                              void* d_out, int out_size, void* d_ws, size_t ws_size,
                              hipStream_t stream) {
  (void)in_sizes; (void)n_in; (void)out_size; (void)ws_size;
  const float* x      = (const float*)d_in[0];
  const float* s      = (const float*)d_in[1];
  const float* W_ih   = (const float*)d_in[2];
  const float* b_ih   = (const float*)d_in[3];
  const float* W_hh   = (const float*)d_in[4];
  const float* b_hh   = (const float*)d_in[5];
  const float* W_out  = (const float*)d_in[6];
  const float* b_out  = (const float*)d_in[7];
  const float* W_halt = (const float*)d_in[8];
  const float* b_halt = (const float*)d_in[9];

  _Float16* Wpack  = (_Float16*)d_ws;
  _Float16* Wopack = (_Float16*)((char*)d_ws + (size_t)KCH * HDIM * 32 * sizeof(_Float16));

  const int total = KCH * HDIM * 32 + OKCH * ODIM * 32;
  prep_weights<<<(total + 255) / 256, 256, 0, stream>>>(W_ih, W_hh, W_out, Wpack, Wopack);

  float* outs = (float*)d_out;
  float* pond = outs + (size_t)S_LEN * BATCH * ODIM;
  act_rnn<<<4, 256, 0, stream>>>(x, s, b_ih, b_hh, b_out, W_halt, b_halt,
                                 Wpack, Wopack, outs, pond);
}
